// ChargeSpinEmbedding_61546881352460
// MI455X (gfx1250) — matrix operand order repaired and checked by reading, not compiled, not run on hardware
//
#include <hip/hip_runtime.h>
#include <hip/hip_bf16.h>

#define N_ATOMS 262144
#define G_GRAPHS 8192
#define F_DIM 128
#define E_ELEM 118
#define EPS 1e-6f
#define INV_SQRT_F 0.08838834764831845f  // 1/sqrt(128)

typedef __attribute__((ext_vector_type(16))) __bf16 v16bf;
typedef __attribute__((ext_vector_type(8)))  __bf16 v8bf;
typedef __attribute__((ext_vector_type(8)))  float  v8f;

__device__ __forceinline__ unsigned short f2bf_bits(float f) {
    union { float f; unsigned u; } c; c.f = f;
    return (unsigned short)(c.u >> 16);          // truncate (bf16 GEMM input)
}
// pack two floats into one bf16x2 dword: clang matches this to v_perm/v_bfi
__device__ __forceinline__ unsigned pack2bf(float lo, float hi) {
    union { float f; unsigned u; } a, b; a.f = lo; b.f = hi;
    return (b.u & 0xFFFF0000u) | (a.u >> 16);
}
__device__ __forceinline__ float silu(float x) {
    // single v_exp_f32 + v_rcp_f32 instead of IEEE-div expansion
    return x * __builtin_amdgcn_rcpf(1.0f + __expf(-x));
}

// ---------------------------------------------------------------------------
// Kernel 0: zero denom, build qk lookup table, convert W1/W2 -> bf16
// ---------------------------------------------------------------------------
__global__ __launch_bounds__(256) void prep_kernel(
    const float* __restrict__ Wq,   // [F,E]
    const float* __restrict__ Wk,   // [2,F]
    const float* __restrict__ W1,   // [F,F]
    const float* __restrict__ W2,   // [F,F]
    float* __restrict__ denom,      // [G]
    float* __restrict__ qkt,        // [2*E] (padded)
    unsigned short* __restrict__ W1b,
    unsigned short* __restrict__ W2b)
{
    int i = blockIdx.x * blockDim.x + threadIdx.x;
    if (i < F_DIM * F_DIM) {
        W1b[i] = f2bf_bits(W1[i]);
        W2b[i] = f2bf_bits(W2[i]);
    }
    if (i < G_GRAPHS) denom[i] = 0.0f;
    if (i < 2 * E_ELEM) {
        int row = i / E_ELEM, e = i % E_ELEM;
        float s = 0.0f;
        #pragma unroll 4
        for (int f = 0; f < F_DIM; ++f)
            s += Wq[f * E_ELEM + e] * Wk[row * F_DIM + f];
        qkt[i] = s;
    }
}

// ---------------------------------------------------------------------------
// Kernel 1: per-atom scalar pass: recover z (argmax of one-hot row),
//           y = softplus(qkt[idx][z]/sqrt(F)), denom += y (segment sum)
// ---------------------------------------------------------------------------
__global__ __launch_bounds__(256) void atoms_kernel(
    const float* __restrict__ one_hot,      // [N,E]
    const float* __restrict__ psi,          // [G]
    const int*   __restrict__ seg,          // [N]
    const float* __restrict__ qkt,          // [2*E]
    float* __restrict__ y_out,              // [N]
    float* __restrict__ denom)              // [G]
{
    int n = blockIdx.x * blockDim.x + threadIdx.x;
    int s = seg[n];
    int idx = (psi[s] < 0.0f) ? 1 : 0;

    // row byte offset = n*472: 8-byte aligned -> float2 loads
    const float2* row2 = (const float2*)(one_hot + (size_t)n * E_ELEM);
    int z = 0; float best = -1.0f;
    #pragma unroll 4
    for (int e2 = 0; e2 < E_ELEM / 2; ++e2) {
        float2 v = row2[e2];
        if (v.x > best) { best = v.x; z = 2 * e2; }
        if (v.y > best) { best = v.y; z = 2 * e2 + 1; }
    }
    float qk = qkt[idx * E_ELEM + z] * INV_SQRT_F;
    float yv = (qk > 20.0f) ? qk : log1pf(__expf(qk));
    y_out[n] = yv;
    atomicAdd(&denom[s], yv);
}

// ---------------------------------------------------------------------------
// Kernel 2: fused att * Wv -> silu -> W1 -> silu -> W2 -> +residual
// 256 threads = 8 wave32; each wave owns a 16-atom x 128-feature tile.
// GEMMs via v_wmma_f32_16x16x32_bf16; inter-layer C->A re-fragmentation
// through a per-wave LDS tile (same-wave DS ordering, s_wait_dscnt).
// ---------------------------------------------------------------------------
__global__ __launch_bounds__(256) void fused_kernel(
    const float* __restrict__ psi,          // [G]
    const int*   __restrict__ seg,          // [N]
    const float* __restrict__ Wv,           // [2,F]
    const float* __restrict__ y_in,         // [N]
    const float* __restrict__ denom,        // [G]
    const unsigned short* __restrict__ W1b, // [F,F] bf16 bits
    const unsigned short* __restrict__ W2b, // [F,F] bf16 bits
    float* __restrict__ out)                // [N,F]
{
    __shared__ float          wv_lds[2 * F_DIM];
    __shared__ float          att_lds[128];
    __shared__ int            idx_lds[128];
    __shared__ unsigned short h_lds[8][16 * F_DIM];   // 8 waves * 4KB

    const int tid  = threadIdx.x;
    const int wave = tid >> 5;
    const int lane = tid & 31;
    const int mrow = lane & 15;      // row (M) within the 16x16 tile
    const int half = lane >> 4;      // lane half selects K sub-group

    for (int i = tid; i < 2 * F_DIM; i += 256) wv_lds[i] = Wv[i];

    const size_t blockBase  = (size_t)blockIdx.x * 128;
    const int    atom_local = wave * 16 + mrow;
    const size_t atom       = blockBase + atom_local;

    // per-atom attention scalar (both lane-halves compute identically)
    int   sg  = seg[atom];
    float ps  = psi[sg];
    int   idx = (ps < 0.0f) ? 1 : 0;
    float att = ps * y_in[atom] / (denom[sg] + EPS);
    if (half == 0) { att_lds[atom_local] = att; idx_lds[atom_local] = idx; }
    __syncthreads();   // wv_lds is cross-wave; att/idx are same-wave

    const float* wvrow = &wv_lds[idx * F_DIM];

    // ---- Layer-1 A fragments built arithmetically: silu(att * Wv[idx][f])
    // A layout (16x32), contiguous K order matching B: lanes<16 hold K = kb*32+{0..15},
    // lanes>=16 hold K = kb*32+{16..31}. Elements built as bf16x2
    // dwords (feature pairs are consecutive and 8B-aligned in LDS).
    v16bf a1[4];
    #pragma unroll
    for (int kb = 0; kb < 4; ++kb) {
        union { unsigned u[8]; v16bf v; } A;
        #pragma unroll
        for (int p = 0; p < 8; ++p) {
            int i0 = 2 * p;
            int f0 = kb * 32 + half * 16 + i0;
            float2 wvp = *(const float2*)(wvrow + f0);   // ds_load_b64
            A.u[p] = pack2bf(silu(att * wvp.x), silu(att * wvp.y));
        }
        a1[kb] = A.v;
    }

    // ---- Layer 1: h1 = silu(v_att) @ W1^T ; stream silu(h1) into LDS tile
    unsigned short* hrow = h_lds[wave];
    #pragma unroll 1
    for (int nt = 0; nt < 8; ++nt) {
        v8f acc = {0.f,0.f,0.f,0.f,0.f,0.f,0.f,0.f};
        #pragma unroll
        for (int kb = 0; kb < 4; ++kb) {
            // B (K x N): lane holds column j = nt*16+mrow,
            // K = kb*32 + half*16 + (0..15) -> 32B contiguous in W1 row j
            const unsigned short* bp =
                W1b + (size_t)(nt * 16 + mrow) * F_DIM + kb * 32 + half * 16;
            v16bf b = *(const v16bf*)bp;
            acc = __builtin_amdgcn_wmma_f32_16x16x32_bf16(
                false, a1[kb], false, b, (short)0, acc, false, false);
        }
        // C layout: VGPR r, lane -> element (M = r + 8*half, Ncol = mrow)
        #pragma unroll
        for (int r = 0; r < 8; ++r) {
            int m = r + 8 * half;
            hrow[m * F_DIM + nt * 16 + mrow] = f2bf_bits(silu(acc[r]));
        }
    }
    // same-wave DS RAW: make the stores visible before transposed reload
    asm volatile("s_wait_dscnt 0x0" ::: "memory");

    // ---- Re-fragment h1 from LDS into A layout for layer 2
    v16bf a2[4];
    #pragma unroll
    for (int kb = 0; kb < 4; ++kb) {
        const unsigned short* p0 = hrow + mrow * F_DIM + kb * 32 + half * 16;
        v8bf lo = *(const v8bf*)p0;          // K = kb*32 + half*16 + 0..7
        v8bf hi = *(const v8bf*)(p0 + 8);    // K = kb*32 + half*16 + 8..15
        a2[kb] = __builtin_shufflevector(lo, hi,
                 0,1,2,3,4,5,6,7,8,9,10,11,12,13,14,15);
    }

    // ---- Layer 2 + residual: out = v_att + silu(h1) @ W2^T
    #pragma unroll 1
    for (int nt = 0; nt < 8; ++nt) {
        v8f acc = {0.f,0.f,0.f,0.f,0.f,0.f,0.f,0.f};
        #pragma unroll
        for (int kb = 0; kb < 4; ++kb) {
            const unsigned short* bp =
                W2b + (size_t)(nt * 16 + mrow) * F_DIM + kb * 32 + half * 16;
            v16bf b = *(const v16bf*)bp;
            acc = __builtin_amdgcn_wmma_f32_16x16x32_bf16(
                false, a2[kb], false, b, (short)0, acc, false, false);
        }
        #pragma unroll
        for (int r = 0; r < 8; ++r) {
            int m  = r + 8 * half;
            float am = att_lds[wave * 16 + m];
            int   im = idx_lds[wave * 16 + m];
            float va = am * wv_lds[im * F_DIM + nt * 16 + mrow];
            out[(blockBase + (size_t)wave * 16 + m) * F_DIM + nt * 16 + mrow]
                = va + acc[r];
        }
    }
}

// ---------------------------------------------------------------------------
extern "C" void kernel_launch(void* const* d_in, const int* in_sizes, int n_in,
                              void* d_out, int out_size, void* d_ws, size_t ws_size,
                              hipStream_t stream) {
    const float* one_hot = (const float*)d_in[0];   // [N,E]
    const float* psi     = (const float*)d_in[1];   // [G]
    const float* Wq      = (const float*)d_in[2];   // [F,E]
    const float* Wk      = (const float*)d_in[3];   // [2,F]
    const float* Wv      = (const float*)d_in[4];   // [2,F]
    const float* W1      = (const float*)d_in[5];   // [F,F]
    const float* W2      = (const float*)d_in[6];   // [F,F]
    const int*   seg     = (const int*)d_in[7];     // [N]
    float* out = (float*)d_out;

    // workspace layout
    float* y     = (float*)d_ws;                    // N floats
    float* denom = y + N_ATOMS;                     // G floats
    float* qkt   = denom + G_GRAPHS;                // 256 floats (pad)
    unsigned short* W1b = (unsigned short*)(qkt + 256);     // F*F bf16
    unsigned short* W2b = W1b + F_DIM * F_DIM;              // F*F bf16

    prep_kernel<<<(F_DIM * F_DIM + 255) / 256, 256, 0, stream>>>(
        Wq, Wk, W1, W2, denom, qkt, W1b, W2b);

    atoms_kernel<<<N_ATOMS / 256, 256, 0, stream>>>(
        one_hot, psi, seg, qkt, y, denom);

    fused_kernel<<<N_ATOMS / 128, 256, 0, stream>>>(
        psi, seg, Wv, y, denom, W1b, W2b, out);
}
